// PerceiverAttention_16458314678373
// MI455X (gfx1250) — matrix operand order repaired and checked by reading, not compiled, not run on hardware
//
#include <hip/hip_runtime.h>
#include <hip/hip_bf16.h>

#define HID   4096
#define HEADS 32
#define KVH   8
#define HD    128
#define QLEN  64
#define CTX   4096
#define KVLEN 4160
#define BSZ   8
#define NCHUNK (KVLEN / 64)   // 65

typedef __attribute__((ext_vector_type(16))) __bf16 v16bf;
typedef __attribute__((ext_vector_type(2)))  __bf16 v2bf;
typedef __attribute__((ext_vector_type(8)))  float  v8f;

union V16 { v16bf v; uint4 q[2]; };

// round-half-up bf16 (1 ulp-tie bias only; cheap: add + shift)
__device__ inline unsigned short f2bf(float f) {
  unsigned int u = __builtin_bit_cast(unsigned int, f) + 0x8000u;
  return (unsigned short)(u >> 16);
}

#if __has_builtin(__builtin_amdgcn_cvt_pk_bf16_f32)
__device__ inline unsigned int pack2(float a, float b) {
  v2bf r = __builtin_amdgcn_cvt_pk_bf16_f32(a, b);
  return __builtin_bit_cast(unsigned int, r);
}
#elif __has_builtin(__builtin_amdgcn_perm)
__device__ inline unsigned int pack2(float a, float b) {
  unsigned int ua = __builtin_bit_cast(unsigned int, a) + 0x8000u;
  unsigned int ub = __builtin_bit_cast(unsigned int, b) + 0x8000u;
  // result = { ub[31:16], ua[31:16] }  (lo16 = a, hi16 = b)
  return __builtin_amdgcn_perm(ub, ua, 0x07060302u);
}
#else
__device__ inline unsigned int pack2(float a, float b) {
  return (unsigned int)f2bf(a) | ((unsigned int)f2bf(b) << 16);
}
#endif

__device__ inline v8f wmma_bf16(v16bf a, v16bf b, v8f c) {
  return __builtin_amdgcn_wmma_f32_16x16x32_bf16(false, a, false, b, (short)0, c, false, false);
}

__device__ inline v16bf ld16(const unsigned short* p) {
  return *reinterpret_cast<const v16bf*>(p);
}

// ---------------------------------------------------------------------------
// bf16-WMMA GEMM, 128(M) x 128(N) workgroup tile, 8 waves, K-step 32,
// software-pipelined with double-buffered LDS (1 barrier / K-step),
// fully templatized so all addressing is compile-time strength-reduced.
// AMODE: 0 = A linear [M][K] fp32; 1 = A rows are concat(context, latents).
// EMODE: 0 = fp32 row-major store
//        1 = RMS-norm per 128-col head + bf16 head-split store
//        3 = bf16 store in attention V B-fragment order (V projection)
// ---------------------------------------------------------------------------
template <int M, int N, int K, int LDB, int AMODE, int EMODE, int RB, int NH>
__global__ __launch_bounds__(256, 1) void pa_gemm(
    const float* __restrict__ A, const float* __restrict__ A2,
    const float* __restrict__ B,
    float* __restrict__ outF, unsigned short* __restrict__ outBF,
    const float* __restrict__ normw)
{
  __shared__ __align__(32) unsigned short ldsA[2][8 * 32 * 16];  // 2 x 8KB
  __shared__ __align__(32) unsigned short ldsB[2][8 * 32 * 16];  // 2 x 8KB
  __shared__ float rowSS[128][2];

  const int tid   = threadIdx.x;
  const int w     = tid >> 5;
  const int lane  = tid & 31;
  const int half  = lane >> 4;
  const int l16   = lane & 15;
  const int p     = w >> 1;     // m-tile pair (0..3)
  const int nhalf = w & 1;      // n-half (4 n-tiles)
  const int m0 = blockIdx.y * 128;
  const int n0 = blockIdx.x * 128;

  // ---- loop-invariant addressing (computed once) ----
  const float* aPtr[4];  int aLds[4];
  #pragma unroll
  for (int j = 0; j < 4; j++) {
    int vi = tid + 256 * j;            // 0..1023 float4s of the A tile
    int am = vi >> 3;                  // 0..127
    int kc = (vi & 7) << 2;            // 0..28 (mult of 4)
    int row = m0 + am;
    const float* aptr;
    if constexpr (AMODE == 1) {
      int b = row / KVLEN;
      int t = row - b * KVLEN;
      aptr = (t < CTX) ? (A  + (size_t)(b * CTX  + t)        * HID)
                       : (A2 + (size_t)(b * QLEN + (t - CTX)) * HID);
    } else {
      aptr = A + (size_t)row * K;
    }
    aPtr[j] = aptr + kc;
    int mtile = am >> 4, mm = am & 15;
    int g = kc >> 4, r = kc & 15, hh = r >> 3, k8 = r & 7;   // k8 in {0,4}
    aLds[j] = ((mtile * 32 + hh * 16 + mm) << 4) + (g * 8 + k8);
  }
  const float* bPtr[4];  int bLds[4];
  #pragma unroll
  for (int j = 0; j < 4; j++) {
    int vi = tid + 256 * j;            // groups of 4 along k
    int n  = vi & 127;
    int bk0 = (vi >> 7) << 2;          // 0,4,..,28
    bPtr[j] = B + (size_t)bk0 * LDB + n0 + n;
    int ntile = n >> 4, nn = n & 15;
    int hh = (bk0 >> 3) & 1, e0 = ((bk0 >> 4) << 3) + (bk0 & 7);
    bLds[j] = ((ntile * 32 + hh * 16 + nn) << 4) + e0;
  }
  const int ldsAF0 = ((p * 2 + 0) * 32 + lane) << 4;
  const int ldsAF1 = ((p * 2 + 1) * 32 + lane) << 4;
  const int ldsBF0 = ((nhalf * 4) * 32 + lane) << 4;

  v8f zf;
  #pragma unroll
  for (int i = 0; i < 8; i++) zf[i] = 0.f;
  v8f acc[2][4];
  #pragma unroll
  for (int i = 0; i < 2; i++)
    #pragma unroll
    for (int nt = 0; nt < 4; nt++) acc[i][nt] = zf;

  float4 ar[4];
  float  br[4][4];

  auto prefetch = [&](int kb) {
    #pragma unroll
    for (int j = 0; j < 4; j++)
      ar[j] = *reinterpret_cast<const float4*>(aPtr[j] + kb);
    #pragma unroll
    for (int j = 0; j < 4; j++) {
      const float* bp = bPtr[j] + (size_t)kb * LDB;
      br[j][0] = bp[0];              br[j][1] = bp[(size_t)LDB];
      br[j][2] = bp[(size_t)2 * LDB]; br[j][3] = bp[(size_t)3 * LDB];
    }
  };
  auto stage = [&](int buf) {
    #pragma unroll
    for (int j = 0; j < 4; j++) {
      uint2 pk; pk.x = pack2(ar[j].x, ar[j].y); pk.y = pack2(ar[j].z, ar[j].w);
      *reinterpret_cast<uint2*>(&ldsA[buf][aLds[j]]) = pk;
    }
    #pragma unroll
    for (int j = 0; j < 4; j++) {
      uint2 pk; pk.x = pack2(br[j][0], br[j][1]); pk.y = pack2(br[j][2], br[j][3]);
      *reinterpret_cast<uint2*>(&ldsB[buf][bLds[j]]) = pk;
    }
  };
  auto compute = [&](int buf) {
    v16bf af0 = ld16(&ldsA[buf][ldsAF0]);
    v16bf af1 = ld16(&ldsA[buf][ldsAF1]);
    #pragma unroll
    for (int nt = 0; nt < 4; nt++) {
      v16bf bfr = ld16(&ldsB[buf][ldsBF0 + (nt * 32 << 4)]);
      acc[0][nt] = wmma_bf16(af0, bfr, acc[0][nt]);
      acc[1][nt] = wmma_bf16(af1, bfr, acc[1][nt]);
    }
  };

  prefetch(0);
  #pragma unroll 1
  for (int kb = 0; kb < K; kb += 64) {          // two pipelined K-steps / iter
    stage(0);
    __syncthreads();
    prefetch(kb + 32);
    compute(0);
    stage(1);
    __syncthreads();
    if (kb + 64 < K) prefetch(kb + 64);
    compute(1);
  }

  if constexpr (EMODE == 0) {
    #pragma unroll
    for (int i = 0; i < 2; i++)
      #pragma unroll
      for (int nt = 0; nt < 4; nt++)
        #pragma unroll
        for (int v = 0; v < 8; v++) {
          int rg = m0 + (p * 2 + i) * 16 + half * 8 + v;
          int cg = n0 + (nhalf * 4 + nt) * 16 + l16;
          outF[(size_t)rg * N + cg] = acc[i][nt][v];
        }
  } else if constexpr (EMODE == 1) {
    // per-row sum of squares over this 128-col block (exactly one head)
    float ss[2][8];
    #pragma unroll
    for (int i = 0; i < 2; i++)
      #pragma unroll
      for (int v = 0; v < 8; v++) {
        float s = 0.f;
        #pragma unroll
        for (int nt = 0; nt < 4; nt++) { float x = acc[i][nt][v]; s += x * x; }
        s += __shfl_xor(s, 1, 32);
        s += __shfl_xor(s, 2, 32);
        s += __shfl_xor(s, 4, 32);
        s += __shfl_xor(s, 8, 32);
        ss[i][v] = s;
      }
    if (l16 == 0) {
      #pragma unroll
      for (int i = 0; i < 2; i++)
        #pragma unroll
        for (int v = 0; v < 8; v++)
          rowSS[(p * 2 + i) * 16 + half * 8 + v][nhalf] = ss[i][v];
    }
    __syncthreads();
    #pragma unroll
    for (int i = 0; i < 2; i++)
      #pragma unroll
      for (int nt = 0; nt < 4; nt++)
        #pragma unroll
        for (int v = 0; v < 8; v++) {
          int rg = m0 + (p * 2 + i) * 16 + half * 8 + v;
          int cg = n0 + (nhalf * 4 + nt) * 16 + l16;
          int d  = cg & (HD - 1);
          int r  = (p * 2 + i) * 16 + half * 8 + v;
          float s2 = rowSS[r][0] + rowSS[r][1];
          float val = acc[i][nt][v] * rsqrtf(s2 * (1.f / 128.f) + 1e-6f) * normw[d];
          int b = rg / RB;
          int t = rg - b * RB;
          int h = cg >> 7;
          size_t idx = (((size_t)(b * NH + h) * RB + t) << 7) + d;
          outBF[idx] = f2bf(val);
        }
  } else {  // EMODE == 3: V projection, store in attention B-fragment order
    #pragma unroll
    for (int i = 0; i < 2; i++)
      #pragma unroll
      for (int nt = 0; nt < 4; nt++)
        #pragma unroll
        for (int v = 0; v < 8; v++) {
          int rg = m0 + (p * 2 + i) * 16 + half * 8 + v;
          int cg = n0 + (nhalf * 4 + nt) * 16 + l16;
          int b   = rg / KVLEN;
          int t   = rg - b * KVLEN;
          int kvh = cg >> 7;
          int d   = cg & (HD - 1);
          int c   = t >> 6, tc = t & 63;
          int kk2 = tc >> 5, rem = tc & 31, hh2 = rem >> 4, e = rem & 15;
          int ntv = d >> 4, nn = d & 15;
          size_t idx =
            ((((((size_t)(b * KVH + kvh) * NCHUNK + c) * 2 + kk2) * 8 + ntv) * 32
              + hh2 * 16 + nn) << 4) + e;
          outBF[idx] = f2bf(acc[i][nt][v]);
        }
  }
}

// ---------------------------------------------------------------------------
// Flash attention: one WAVE (32 threads) per (b, h, 16-query strip).
// kv chunks of 64. K and V fragments load straight from global (V is stored
// in B-fragment order by the projection). Only P round-trips through LDS.
// ---------------------------------------------------------------------------
__global__ __launch_bounds__(32, 1) void pa_attn(
    const unsigned short* __restrict__ Qn,
    const unsigned short* __restrict__ Kn,
    const unsigned short* __restrict__ Vn,
    float* __restrict__ Ao)
{
  __shared__ __align__(32) unsigned short ldsP[16][64];

  const int lane = threadIdx.x;
  const int half = lane >> 4;
  const int l16  = lane & 15;
  const int h    = blockIdx.x;
  const int b    = blockIdx.y;
  const int mt   = blockIdx.z;              // 16-query strip (0..3)
  const int kvh  = h >> 2;
  const size_t kvrow0 = (size_t)(b * KVH + kvh) * KVLEN;
  const size_t vbase0 = (size_t)(b * KVH + kvh) * NCHUNK;

  // Preload Q A-fragments (K=128 -> 4 fragments), direct from global.
  const unsigned short* qbase =
      Qn + ((size_t)(b * HEADS + h) * QLEN + mt * 16 + l16) * HD;
  V16 aq[4];
  #pragma unroll
  for (int kk = 0; kk < 4; kk++) {
    int k0 = kk * 32 + half * 16;
    aq[kk].q[0] = *reinterpret_cast<const uint4*>(qbase + k0);
    aq[kk].q[1] = *reinterpret_cast<const uint4*>(qbase + k0 + 8);
  }

  v8f zf;
  #pragma unroll
  for (int i = 0; i < 8; i++) zf[i] = 0.f;
  v8f oacc[8];
  #pragma unroll
  for (int t = 0; t < 8; t++) oacc[t] = zf;
  float rmax[8], rsum[8];
  #pragma unroll
  for (int v = 0; v < 8; v++) { rmax[v] = -3e30f; rsum[v] = 0.f; }

  const float sc = 0.08838834764831845f;   // 1/sqrt(128)

  #pragma unroll 1
  for (int ch = 0; ch < NCHUNK; ch++) {
    const int t0 = ch * 64;
    // ---- S = Q K^T for this 16x64 strip (K frags contiguous in global) ----
    v8f sacc[4] = {zf, zf, zf, zf};
    #pragma unroll
    for (int nt = 0; nt < 4; nt++) {
      const unsigned short* kp0 = Kn + (kvrow0 + t0 + nt * 16 + l16) * HD + half * 16;
      #pragma unroll
      for (int kk = 0; kk < 4; kk++) {
        v16bf bk = ld16(kp0 + kk * 32);
        sacc[nt] = wmma_bf16(aq[kk].v, bk, sacc[nt]);
      }
    }
    // ---- online softmax ----
    float cf[8], csum[8];
    #pragma unroll
    for (int v = 0; v < 8; v++) {
      float m = fmaxf(fmaxf(sacc[0][v], sacc[1][v]), fmaxf(sacc[2][v], sacc[3][v]));
      m = fmaxf(m, __shfl_xor(m, 1, 32));
      m = fmaxf(m, __shfl_xor(m, 2, 32));
      m = fmaxf(m, __shfl_xor(m, 4, 32));
      m = fmaxf(m, __shfl_xor(m, 8, 32));
      float nm = fmaxf(rmax[v], m * sc);
      cf[v] = __expf(rmax[v] - nm);
      rmax[v] = nm;
      csum[v] = 0.f;
    }
    __syncthreads();   // single-wave WG (-> S_NOP) but keeps LDS ordering
    #pragma unroll
    for (int nt = 0; nt < 4; nt++)
      #pragma unroll
      for (int v = 0; v < 8; v++) {
        float pp = __expf(sacc[nt][v] * sc - rmax[v]);
        csum[v] += pp;
        ldsP[half * 8 + v][nt * 16 + l16] = f2bf(pp);  // C-layout -> A-layout
      }
    #pragma unroll
    for (int v = 0; v < 8; v++) {
      float s = csum[v];
      s += __shfl_xor(s, 1, 32);
      s += __shfl_xor(s, 2, 32);
      s += __shfl_xor(s, 4, 32);
      s += __shfl_xor(s, 8, 32);
      rsum[v] = rsum[v] * cf[v] + s;
    }
    #pragma unroll
    for (int nt2 = 0; nt2 < 8; nt2++)
      #pragma unroll
      for (int v = 0; v < 8; v++) oacc[nt2][v] *= cf[v];
    __syncthreads();
    // ---- O += P * V  (V frags contiguous in global, pre-laid-out) ----
    #pragma unroll
    for (int kk2 = 0; kk2 < 2; kk2++) {
      V16 ap;
      int k0 = kk2 * 32 + half * 16;
      ap.q[0] = *reinterpret_cast<const uint4*>(&ldsP[l16][k0]);
      ap.q[1] = *reinterpret_cast<const uint4*>(&ldsP[l16][k0 + 8]);
      const unsigned short* vb =
          Vn + ((((vbase0 + ch) * 2 + kk2) * 8) * 32 << 4) + (lane << 4);
      #pragma unroll
      for (int nt2 = 0; nt2 < 8; nt2++) {
        v16bf bv = ld16(vb + (nt2 * 32 << 4));
        oacc[nt2] = wmma_bf16(ap.v, bv, oacc[nt2]);
      }
    }
  }

  // ---- normalize and store fp32 [b][t][h*128+d] ----
  #pragma unroll
  for (int v = 0; v < 8; v++) {
    float inv = 1.f / rsum[v];
    int row = mt * 16 + half * 8 + v;
    #pragma unroll
    for (int nt2 = 0; nt2 < 8; nt2++) {
      int d = nt2 * 16 + l16;
      Ao[(size_t)(b * QLEN + row) * (HEADS * HD) + h * HD + d] = oacc[nt2][v] * inv;
    }
  }
}

extern "C" void kernel_launch(void* const* d_in, const int* in_sizes, int n_in,
                              void* d_out, int out_size, void* d_ws, size_t ws_size,
                              hipStream_t stream) {
  const float* latents = (const float*)d_in[0];
  const float* context = (const float*)d_in[1];
  const float* Wq = (const float*)d_in[2];
  const float* Wk = (const float*)d_in[3];
  const float* Wv = (const float*)d_in[4];
  const float* Wo = (const float*)d_in[5];
  const float* qnw = (const float*)d_in[6];
  const float* knw = (const float*)d_in[7];

  // workspace layout (bf16 Qn/Kn/Vn, fp32 attention output) ~ 149 MB
  unsigned short* Qn = (unsigned short*)d_ws;
  unsigned short* Kn = Qn + (size_t)BSZ * HEADS * QLEN * HD;
  unsigned short* Vn = Kn + (size_t)BSZ * KVH * KVLEN * HD;
  float*          Ao = (float*)(Vn + (size_t)BSZ * KVH * KVLEN * HD);

  const int Mq  = BSZ * QLEN;    // 512
  const int Mkv = BSZ * KVLEN;   // 33280

  // Q projection + q_norm -> bf16 [b][h][64][128]
  pa_gemm<512, 4096, 4096, 4096, 0, 1, QLEN, HEADS>
      <<<dim3(4096 / 128, Mq / 128), 256, 0, stream>>>(
          latents, nullptr, Wq, nullptr, Qn, qnw);
  // K projection (concat rows) + k_norm -> bf16 [b][kvh][4160][128]
  pa_gemm<33280, 1024, 4096, 1024, 1, 1, KVLEN, KVH>
      <<<dim3(1024 / 128, Mkv / 128), 256, 0, stream>>>(
          context, latents, Wk, nullptr, Kn, knw);
  // V projection (concat rows) -> bf16 in attention B-fragment order
  pa_gemm<33280, 1024, 4096, 1024, 1, 3, KVLEN, KVH>
      <<<dim3(1024 / 128, Mkv / 128), 256, 0, stream>>>(
          context, latents, Wv, nullptr, Vn, nullptr);
  // attention: one wave per (b, h, 16-query strip)
  pa_attn<<<dim3(HEADS, BSZ, 4), 32, 0, stream>>>(Qn, Kn, Vn, Ao);
  // output projection -> fp32 d_out [8][64][4096]
  pa_gemm<512, 4096, 4096, 4096, 0, 0, QLEN, 1>
      <<<dim3(4096 / 128, Mq / 128), 256, 0, stream>>>(
          Ao, nullptr, Wo, (float*)d_out, nullptr, nullptr);
}
